// SwinTransformer_mona_65180423685493
// MI455X (gfx1250) — matrix operand order repaired and checked by reading, not compiled, not run on hardware
//
#include <hip/hip_runtime.h>
#include <math.h>

typedef _Float16 half_t;
typedef __attribute__((ext_vector_type(16))) _Float16 v16h;
typedef __attribute__((ext_vector_type(8)))  float    v8f;

#define DEVI __device__ __forceinline__

namespace {
constexpr int kB = 16, kH = 56, kWid = 56, kC = 384;
constexpr int kWS = 7, kSS = 3, kNH = 12;
constexpr int kN = 49, kNWIN = 64;                 // tokens/window, windows/image
constexpr int kL = kH * kWid;                      // 3136
constexpr long long kBL = (long long)kB * kL;      // 50176 rows
constexpr int kHID = 4 * kC;                       // 1536
constexpr int kMH = 64;                            // Mona hidden
}

// ---------------- WMMA fragment loaders (ISA 7.12.2 layouts, wave32) ----------
// A (16x32 f16): lane m<16 -> row m, K {k0..k0+7, k0+16..k0+23}
//                lane m+16 -> row m, K {k0+8..k0+15, k0+24..k0+31}
DEVI v16h load_a_frag(const half_t* A, int lda, int m0, int k0) {
  int lane = threadIdx.x & 31;
  int r = lane & 15, hi = lane >> 4;
  const half_t* p = A + (size_t)(m0 + r) * lda + (k0 + hi * 8);
  union { v16h v; uint4 u[2]; } f;
  f.u[0] = *(const uint4*)(p);
  f.u[1] = *(const uint4*)(p + 16);
  return f.v;
}
// B (32x16 f16) from row-major W[N,K]: lane n<16 -> col n, K {k0..k0+7, k0+16..k0+23};
//                                      lane n+16 -> col n, K {k0+8..k0+15, k0+24..k0+31}
DEVI v16h load_b_frag(const half_t* W, int ldk, int n0, int k0) {
  int lane = threadIdx.x & 31;
  int n = lane & 15, hi = lane >> 4;
  const half_t* p = W + (size_t)(n0 + n) * ldk + (k0 + hi * 8);
  union { v16h v; uint4 u[2]; } f;
  f.u[0] = *(const uint4*)(p);
  f.u[1] = *(const uint4*)(p + 16);
  return f.v;
}
DEVI v8f wmma16(v16h a, v16h b, v8f c) {
  return __builtin_amdgcn_wmma_f32_16x16x32_f16(false, a, false, b, (short)0, c, false, false);
}
DEVI float gelu_exact(float x) { return 0.5f * x * (1.0f + erff(x * 0.70710678118654752f)); }

// ---------------- utility kernels ---------------------------------------------
__global__ void cvt_f2h_kernel(const float* __restrict__ s, half_t* __restrict__ d, long long n) {
  long long i = (long long)blockIdx.x * blockDim.x + threadIdx.x;
  long long st = (long long)gridDim.x * blockDim.x;
  for (; i < n; i += st) d[i] = (half_t)s[i];
}

// generic LayerNorm over C=384, f32 in -> f16 out (same row order)
__global__ void ln_rows_kernel(const float* __restrict__ x, const float* __restrict__ g,
                               const float* __restrict__ bt, half_t* __restrict__ out) {
  int row = blockIdx.x, tid = threadIdx.x;
  const float* p = x + (size_t)row * kC;
  float v[3], s = 0.f, s2 = 0.f;
#pragma unroll
  for (int i = 0; i < 3; ++i) { v[i] = p[tid + i * 128]; s += v[i]; s2 += v[i] * v[i]; }
  for (int off = 16; off > 0; off >>= 1) { s += __shfl_xor(s, off); s2 += __shfl_xor(s2, off); }
  __shared__ float rs[4], rq[4];
  int wv = tid >> 5;
  if ((tid & 31) == 0) { rs[wv] = s; rq[wv] = s2; }
  __syncthreads();
  s = rs[0] + rs[1] + rs[2] + rs[3];
  s2 = rq[0] + rq[1] + rq[2] + rq[3];
  float mean = s * (1.f / kC);
  float var = s2 * (1.f / kC) - mean * mean;
  float rstd = rsqrtf(var + 1e-5f);
  half_t* o = out + (size_t)row * kC;
#pragma unroll
  for (int i = 0; i < 3; ++i) {
    int c = tid + i * 128;
    o[c] = (half_t)((v[i] - mean) * rstd * g[c] + bt[c]);
  }
}

// LN1 fused with cyclic shift (-3,-3) + window partition: out row = windowed order
__global__ void ln1_window_kernel(const float* __restrict__ x, const float* __restrict__ g,
                                  const float* __restrict__ bt, half_t* __restrict__ out) {
  int r = blockIdx.x, tid = threadIdx.x;
  int b = r / (kNWIN * kN);
  int rem = r % (kNWIN * kN);
  int win = rem / kN, pos = rem % kN;
  int sh = (win / 8) * kWS + pos / kWS;
  int sw = (win % 8) * kWS + pos % kWS;
  int h = (sh + kSS) % kH, w = (sw + kSS) % kWid;
  const float* p = x + ((size_t)b * kL + h * kWid + w) * kC;
  float v[3], s = 0.f, s2 = 0.f;
#pragma unroll
  for (int i = 0; i < 3; ++i) { v[i] = p[tid + i * 128]; s += v[i]; s2 += v[i] * v[i]; }
  for (int off = 16; off > 0; off >>= 1) { s += __shfl_xor(s, off); s2 += __shfl_xor(s2, off); }
  __shared__ float rs[4], rq[4];
  int wv = tid >> 5;
  if ((tid & 31) == 0) { rs[wv] = s; rq[wv] = s2; }
  __syncthreads();
  s = rs[0] + rs[1] + rs[2] + rs[3];
  s2 = rq[0] + rq[1] + rq[2] + rq[3];
  float mean = s * (1.f / kC);
  float rstd = rsqrtf(s2 * (1.f / kC) - mean * mean + 1e-5f);
  half_t* o = out + (size_t)r * kC;
#pragma unroll
  for (int i = 0; i < 3; ++i) {
    int c = tid + i * 128;
    o[c] = (half_t)((v[i] - mean) * rstd * g[c] + bt[c]);
  }
}

// ---------------- generic WMMA GEMM: out[M,N] = A[M,K] @ W[N,K]^T + bias ------
// block = 128x64 tile, 256 threads (8 waves, 4Mx2N), each wave: 32x32 (2x2 tiles)
// -> per K-step: 4 frag loads feed 4 WMMAs (A and B frags both reused twice).
// mode 0: f16 out; 1: f16 gelu; 2: f32 out; 3: f32 out + res; 4: res + out*gamma
__global__ void gemm_wmma_kernel(const half_t* __restrict__ A, const half_t* __restrict__ Wt,
                                 const float* __restrict__ bias, int K, int mode,
                                 half_t* __restrict__ outh, float* __restrict__ outf,
                                 const float* __restrict__ res, const float* __restrict__ gamma,
                                 int ldo) {
  int tid = threadIdx.x, wv = tid >> 5;
  int m0 = blockIdx.x * 128 + (wv >> 1) * 32;
  int n0 = blockIdx.y * 64 + (wv & 1) * 32;
  v8f acc00 = {}, acc01 = {}, acc10 = {}, acc11 = {};
  for (int k0 = 0; k0 < K; k0 += 32) {
    v16h a0 = load_a_frag(A, K, m0, k0);
    v16h a1 = load_a_frag(A, K, m0 + 16, k0);
    v16h b0 = load_b_frag(Wt, K, n0, k0);
    v16h b1 = load_b_frag(Wt, K, n0 + 16, k0);
    acc00 = wmma16(a0, b0, acc00);
    acc01 = wmma16(a0, b1, acc01);
    acc10 = wmma16(a1, b0, acc10);
    acc11 = wmma16(a1, b1, acc11);
  }
  int lane = tid & 31, cn = lane & 15, hi = lane >> 4;
#pragma unroll
  for (int mi = 0; mi < 2; ++mi) {
#pragma unroll
    for (int r = 0; r < 8; ++r) {
      int row = m0 + mi * 16 + r + hi * 8;
#pragma unroll
      for (int t = 0; t < 2; ++t) {
        int col = n0 + t * 16 + cn;
        float v = (mi ? (t ? acc11[r] : acc10[r]) : (t ? acc01[r] : acc00[r])) + bias[col];
        size_t o = (size_t)row * ldo + col;
        if (mode == 0)      outh[o] = (half_t)v;
        else if (mode == 1) outh[o] = (half_t)gelu_exact(v);
        else if (mode == 2) outf[o] = v;
        else if (mode == 3) outf[o] = v + res[o];
        else                outf[o] = res[o] + v * gamma[col];
      }
    }
  }
}

// ---------------- windowed attention: one block per (window, head) ------------
__global__ void attn_kernel(const half_t* __restrict__ qkv, const float* __restrict__ rpb,
                            const float* __restrict__ mask, half_t* __restrict__ out) {
  int blk = blockIdx.x;
  int w = blk / kNH, head = blk % kNH;
  int wi = w % kNWIN;
  __shared__ __align__(16) half_t q_s[64][32];
  __shared__ __align__(16) half_t k_s[64][32];
  __shared__ __align__(16) half_t vT_s[32][64];
  __shared__ __align__(16) float  S_s[64][64];
  __shared__ __align__(16) half_t P_s[64][64];
  int tid = threadIdx.x;
  size_t base = (size_t)w * kN * (3 * kC);
  for (int idx = tid; idx < 64 * 32; idx += 256) {
    int t = idx >> 5, d = idx & 31;
    float qv = 0.f, kv = 0.f, vv = 0.f;
    if (t < kN) {
      const half_t* row = qkv + base + (size_t)t * (3 * kC);
      qv = (float)row[head * 32 + d] * 0.17677669529663687f;  // hd^-0.5
      kv = (float)row[kC + head * 32 + d];
      vv = (float)row[2 * kC + head * 32 + d];
    }
    q_s[t][d] = (half_t)qv;
    k_s[t][d] = (half_t)kv;
    vT_s[d][t] = (half_t)vv;
  }
  __syncthreads();
  int wv = tid >> 5, lane = tid & 31, cn = lane & 15, hi = lane >> 4;
  // S = (q*scale) @ k^T : 64x64, K=32 -> one WMMA per tile, 16 tiles, 2 per wave
#pragma unroll
  for (int p = 0; p < 2; ++p) {
    int tix = wv * 2 + p;
    int mt = tix >> 2, nt = tix & 3;
    v16h a = load_a_frag(&q_s[0][0], 32, mt * 16, 0);
    v16h b = load_b_frag(&k_s[0][0], 32, nt * 16, 0);
    v8f acc = {};
    acc = wmma16(a, b, acc);
#pragma unroll
    for (int r = 0; r < 8; ++r) S_s[mt * 16 + r + hi * 8][nt * 16 + cn] = acc[r];
  }
  __syncthreads();
  // rpb + mask + softmax: thread n owns row n (n < 64)
  if (tid < 64) {
    int n = tid;
    if (n < kN) {
      int i1 = n / 7, j1 = n % 7;
      float mx = -3.0e38f;
      for (int m = 0; m < 64; ++m) {
        float val = -3.0e38f;
        if (m < kN) {
          int i2 = m / 7, j2 = m % 7;
          int rel = (i1 - i2 + 6) * 13 + (j1 - j2 + 6);
          val = S_s[n][m] + rpb[rel * kNH + head] + mask[(size_t)wi * kN * kN + n * kN + m];
        }
        S_s[n][m] = val;
        mx = fmaxf(mx, val);
      }
      float sum = 0.f;
      for (int m = 0; m < 64; ++m) {
        float e = (m < kN) ? __expf(S_s[n][m] - mx) : 0.f;
        S_s[n][m] = e;
        sum += e;
      }
      float inv = 1.f / sum;
      for (int m = 0; m < 64; ++m) P_s[n][m] = (half_t)(S_s[n][m] * inv);
    } else {
      for (int m = 0; m < 64; ++m) P_s[n][m] = (half_t)0.f;
    }
  }
  __syncthreads();
  // O = P @ V : 64x32, K=64 -> 8 tiles, one per wave, 2 WMMA each
  int mt = wv >> 1, nt = wv & 1;
  v8f acc = {};
  for (int k0 = 0; k0 < 64; k0 += 32) {
    v16h a = load_a_frag(&P_s[0][0], 64, mt * 16, k0);
    v16h b = load_b_frag(&vT_s[0][0], 64, nt * 16, k0);
    acc = wmma16(a, b, acc);
  }
#pragma unroll
  for (int r = 0; r < 8; ++r) {
    int t = mt * 16 + r + hi * 8;
    if (t < kN)
      out[((size_t)w * kN + t) * kC + head * 32 + nt * 16 + cn] = (half_t)acc[r];
  }
}

// reverse window partition + reverse shift (+3,+3) + residual add with input x
__global__ void unwindow_res_kernel(const float* __restrict__ x, const float* __restrict__ pw,
                                    float* __restrict__ o) {
  long long n = kBL * kC;
  long long i = (long long)blockIdx.x * blockDim.x + threadIdx.x;
  long long st = (long long)gridDim.x * blockDim.x;
  for (; i < n; i += st) {
    long long t = i / kC;
    int c = (int)(i % kC);
    int b = (int)(t / kL);
    int pos = (int)(t % kL);
    int h = pos / kWid, w = pos % kWid;
    int sh = (h + kH - kSS) % kH, sw = (w + kWid - kSS) % kWid;
    long long r = (long long)b * (kNWIN * kN) + ((sh / kWS) * 8 + (sw / kWS)) * kN +
                  (sh % kWS) * kWS + (sw % kWS);
    o[i] = x[i] + pw[r * kC + c];
  }
}

// Mona depthwise 3x3+5x5+7x7 (avg of three) + identity, NHWC f16 in, f32+f16 out
__global__ void mona_dwconv_kernel(const half_t* __restrict__ h1,
                                   const float* __restrict__ w3, const float* __restrict__ b3,
                                   const float* __restrict__ w5, const float* __restrict__ b5,
                                   const float* __restrict__ w7, const float* __restrict__ b7,
                                   float* __restrict__ c0f, half_t* __restrict__ c0h) {
  int bh = blockIdx.x;
  int b = bh / kH, h = bh % kH;
  for (int idx = threadIdx.x; idx < kWid * kMH; idx += 256) {
    int w = idx >> 6, ch = idx & 63;
    float acc = b3[ch] + b5[ch] + b7[ch];
    float center = 0.f;
    for (int dy = -3; dy <= 3; ++dy) {
      int y = h + dy;
      if (y < 0 || y >= kH) continue;
      for (int dx = -3; dx <= 3; ++dx) {
        int x = w + dx;
        if (x < 0 || x >= kWid) continue;
        float p = (float)h1[(size_t)(b * kL + y * kWid + x) * kMH + ch];
        acc += p * w7[ch * 49 + (dy + 3) * 7 + (dx + 3)];
        if (dy >= -2 && dy <= 2 && dx >= -2 && dx <= 2)
          acc += p * w5[ch * 25 + (dy + 2) * 5 + (dx + 2)];
        if (dy >= -1 && dy <= 1 && dx >= -1 && dx <= 1) {
          acc += p * w3[ch * 9 + (dy + 1) * 3 + (dx + 1)];
          if (dy == 0 && dx == 0) center = p;
        }
      }
    }
    float r = acc * (1.f / 3.f) + center;
    size_t o = (size_t)(b * kL + h * kWid + w) * kMH + ch;
    c0f[o] = r;
    c0h[o] = (half_t)r;
  }
}

// h2 = gelu(c0 + proj1x1(c0))
__global__ void add_gelu_kernel(const float* __restrict__ a, const float* __restrict__ b,
                                half_t* __restrict__ o, long long n) {
  long long i = (long long)blockIdx.x * blockDim.x + threadIdx.x;
  long long st = (long long)gridDim.x * blockDim.x;
  for (; i < n; i += st) o[i] = (half_t)gelu_exact(a[i] + b[i]);
}

// ---------------- host orchestration ------------------------------------------
extern "C" void kernel_launch(void* const* d_in, const int* in_sizes, int n_in,
                              void* d_out, int out_size, void* d_ws, size_t ws_size,
                              hipStream_t stream) {
  (void)in_sizes; (void)n_in; (void)out_size; (void)ws_size;
  const float* x      = (const float*)d_in[0];
  const float* mask   = (const float*)d_in[1];
  const float* n1g    = (const float*)d_in[2];
  const float* n1b    = (const float*)d_in[3];
  const float* qkv_w  = (const float*)d_in[4];
  const float* qkv_b  = (const float*)d_in[5];
  const float* rpb    = (const float*)d_in[6];
  const float* proj_w = (const float*)d_in[7];
  const float* proj_b = (const float*)d_in[8];
  const float* n2g    = (const float*)d_in[9];
  const float* n2b    = (const float*)d_in[10];
  const float* fc1_w  = (const float*)d_in[11];
  const float* fc1_b  = (const float*)d_in[12];
  const float* fc2_w  = (const float*)d_in[13];
  const float* fc2_b  = (const float*)d_in[14];
  // mona params: 0 norm_g 1 norm_b 2 p1_w 3 p1_b 4 c1_w 5 c1_b 6 c2_w 7 c2_b
  //              8 c3_w 9 c3_b 10 proj_w 11 proj_b 12 p2_w 13 p2_b 14 gamma
  const float* mp[2][15];
  for (int m = 0; m < 2; ++m)
    for (int i = 0; i < 15; ++i) mp[m][i] = (const float*)d_in[15 + m * 15 + i];

  char* ws = (char*)d_ws;
  size_t off = 0;
  auto alloc = [&](size_t bytes) {
    size_t o = off;
    off = (off + bytes + 255) & ~(size_t)255;
    return o;
  };
  // fp16 weights
  half_t* wqkv_h = (half_t*)(ws + alloc((size_t)3 * kC * kC * 2));
  half_t* wprj_h = (half_t*)(ws + alloc((size_t)kC * kC * 2));
  half_t* wfc1_h = (half_t*)(ws + alloc((size_t)kHID * kC * 2));
  half_t* wfc2_h = (half_t*)(ws + alloc((size_t)kC * kHID * 2));
  half_t* mw_p1[2]; half_t* mw_pj[2]; half_t* mw_p2[2];
  for (int m = 0; m < 2; ++m) {
    mw_p1[m] = (half_t*)(ws + alloc((size_t)kMH * kC * 2));
    mw_pj[m] = (half_t*)(ws + alloc((size_t)kMH * kMH * 2));
    mw_p2[m] = (half_t*)(ws + alloc((size_t)kC * kMH * 2));
  }
  // reused activation regions
  char* regA = ws + alloc((size_t)kBL * kHID * 2);   // qkv_h / fc1_h (154 MB)
  char* regB = ws + alloc((size_t)kBL * kC * 2);     // ln halves / attn out (39 MB)
  char* regC = ws + alloc((size_t)kBL * kC * 4);     // projout / projc / x2 (77 MB)
  char* regD = ws + alloc((size_t)kBL * kC * 4);     // x1 / x3 (77 MB)
  char* regE = ws + alloc((size_t)kBL * kMH * 2);    // h1 / h2 (6.4 MB)
  char* regF = ws + alloc((size_t)kBL * kMH * 4);    // c0 f32 (12.8 MB)
  char* regG = ws + alloc((size_t)kBL * kMH * 2);    // c0 f16 (6.4 MB)

  auto cvt = [&](const float* s, half_t* d, long long n) {
    unsigned g = (unsigned)((n + 256 * 8 - 1) / (256 * 8));
    cvt_f2h_kernel<<<dim3(g), dim3(256), 0, stream>>>(s, d, n);
  };
  auto gemm = [&](const half_t* A, const half_t* Wt, const float* bias, long long M, int Nn,
                  int K, int mode, half_t* oh, float* of, const float* res, const float* gm) {
    gemm_wmma_kernel<<<dim3((unsigned)(M / 128), (unsigned)(Nn / 64)), dim3(256), 0, stream>>>(
        A, Wt, bias, K, mode, oh, of, res, gm, Nn);
  };

  // ---- weight conversion (every launch; deterministic) ----
  cvt(qkv_w, wqkv_h, (long long)3 * kC * kC);
  cvt(proj_w, wprj_h, (long long)kC * kC);
  cvt(fc1_w, wfc1_h, (long long)kHID * kC);
  cvt(fc2_w, wfc2_h, (long long)kC * kHID);
  for (int m = 0; m < 2; ++m) {
    cvt(mp[m][2], mw_p1[m], (long long)kMH * kC);
    cvt(mp[m][10], mw_pj[m], (long long)kMH * kMH);
    cvt(mp[m][12], mw_p2[m], (long long)kC * kMH);
  }

  half_t* lnh  = (half_t*)regB;
  half_t* qkvh = (half_t*)regA;

  // ---- attention branch ----
  ln1_window_kernel<<<dim3((unsigned)kBL), dim3(128), 0, stream>>>(x, n1g, n1b, lnh);
  gemm(lnh, wqkv_h, qkv_b, kBL, 3 * kC, kC, 0, qkvh, nullptr, nullptr, nullptr);
  half_t* attnh = (half_t*)regB;  // ln consumed
  attn_kernel<<<dim3((unsigned)(kB * kNWIN * kNH)), dim3(256), 0, stream>>>(qkvh, rpb, mask, attnh);
  float* projwin = (float*)regC;
  gemm(attnh, wprj_h, proj_b, kBL, kC, kC, 2, nullptr, projwin, nullptr, nullptr);
  float* x1 = (float*)regD;
  unwindow_res_kernel<<<dim3(4096), dim3(256), 0, stream>>>(x, projwin, x1);

  // ---- mona adapter (shared routine) ----
  auto mona = [&](int m, const float* xin, float* xout) {
    half_t* lh = (half_t*)regB;
    ln_rows_kernel<<<dim3((unsigned)kBL), dim3(128), 0, stream>>>(xin, mp[m][0], mp[m][1], lh);
    half_t* h1 = (half_t*)regE;
    gemm(lh, mw_p1[m], mp[m][3], kBL, kMH, kC, 0, h1, nullptr, nullptr, nullptr);
    float* c0f = (float*)regF;
    half_t* c0h = (half_t*)regG;
    mona_dwconv_kernel<<<dim3((unsigned)(kB * kH)), dim3(256), 0, stream>>>(
        h1, mp[m][4], mp[m][5], mp[m][6], mp[m][7], mp[m][8], mp[m][9], c0f, c0h);
    float* prc = (float*)regC;
    gemm(c0h, mw_pj[m], mp[m][11], kBL, kMH, kMH, 2, nullptr, prc, nullptr, nullptr);
    half_t* h2 = (half_t*)regE;  // h1 consumed
    add_gelu_kernel<<<dim3(2048), dim3(256), 0, stream>>>(c0f, prc, h2, kBL * kMH);
    gemm(h2, mw_p2[m], mp[m][13], kBL, kC, kMH, 4, nullptr, xout, xin, mp[m][14]);
  };

  float* x2 = (float*)regC;  // projc region reused after it is consumed inside mona
  mona(0, x1, x2);

  // ---- MLP ----
  ln_rows_kernel<<<dim3((unsigned)kBL), dim3(128), 0, stream>>>(x2, n2g, n2b, lnh);
  half_t* fc1h = (half_t*)regA;  // qkv consumed
  gemm(lnh, wfc1_h, fc1_b, kBL, kHID, kC, 1, fc1h, nullptr, nullptr, nullptr);
  float* x3 = (float*)regD;  // x1 consumed
  gemm(fc1h, wfc2_h, fc2_b, kBL, kC, kHID, 3, nullptr, x3, x2, nullptr);

  // ---- mona2 -> final output ----
  mona(1, x3, (float*)d_out);
}
